// Generator_89713276878905
// MI455X (gfx1250) — compile-verified, matrix operand order checked by reading, not run on hardware
//
#include <hip/hip_runtime.h>
#include <hip/hip_bf16.h>
#include <math.h>
#include <stdint.h>

typedef __attribute__((ext_vector_type(16))) _Float16 v16h;
typedef __attribute__((ext_vector_type(8)))  float    v8f;
typedef __attribute__((ext_vector_type(4)))  int      v4i;

// address-space-qualified pointers for the async-to-LDS builtin
typedef __attribute__((address_space(1))) v4i* as1_v4i_p;   // global
typedef __attribute__((address_space(3))) v4i* as3_v4i_p;   // LDS

#define N_NODES 8192
#define C_DIM   256
#define H_DIM   128
#define B_G     32
#define NCLS_   2
#define E_EDGES 131072

// GEMM tiling: 128x128 block tile, K-step 32 (one WMMA f16 K), 256 threads = 8 waves (2x4)
#define BM 128
#define BN 128
#define BK 32
#define SA   (BK + 8)    // f16 LDS row stride (halfs)
#define SB   (BN + 8)
#define SA32 (BK + 4)    // f32 LDS row stride (floats), multiple of 4 for b128 stores
#define SPLITK 8
#define KSLICE (N_NODES / SPLITK)   // 1024

#if __has_builtin(__builtin_amdgcn_global_load_async_to_lds_b128)
#define ASYNC_OK 1
#else
#define ASYNC_OK 0
#endif

// ---------------------------------------------------------------------------
// Fragment layouts per CDNA5 ISA 7.12.2 (f16 WMMA 16x16x32, wave32):
//   A: lane<16 -> M=lane, v0..3:K0..7, v4..7:K16..23 ; lane>=16 -> +8 on both
//   B: mirrored, lane%16 selects N column
//   C: VGPR r, lane l -> M = r + 8*(l>=16), N = l%16
// ---------------------------------------------------------------------------
__device__ __forceinline__ int frag_kb(int p, int hh) {
  return (p < 4) ? (8 * hh + 2 * p) : (16 + 8 * hh + 2 * (p - 4));
}

__device__ __forceinline__ void load_bfrag(const _Float16 (&Bs)[BK][SB],
                                           v16h (&bf)[2], int wc, int hh, int mr) {
#pragma unroll
  for (int j = 0; j < 2; ++j) {
    const int n = wc * 32 + j * 16 + mr;
#pragma unroll
    for (int p = 0; p < 8; ++p) {
      const int kb = frag_kb(p, hh);
      bf[j][2 * p]     = Bs[kb][n];
      bf[j][2 * p + 1] = Bs[kb + 1][n];
    }
  }
}

// MMA step, A held as f16 in LDS
__device__ __forceinline__ void mma_k_f16(const _Float16 (&As)[BM][SA],
                                          const _Float16 (&Bs)[BK][SB],
                                          v8f (&acc)[4][2],
                                          int wr, int wc, int hh, int mr) {
  v16h bf[2];
  load_bfrag(Bs, bf, wc, hh, mr);
#pragma unroll
  for (int i = 0; i < 4; ++i) {
    v16h af;
    const int m = wr * 64 + i * 16 + mr;
#pragma unroll
    for (int p = 0; p < 8; ++p) {
      const int kb = frag_kb(p, hh);
      af[2 * p]     = As[m][kb];
      af[2 * p + 1] = As[m][kb + 1];
    }
#pragma unroll
    for (int j = 0; j < 2; ++j)
      acc[i][j] = __builtin_amdgcn_wmma_f32_16x16x32_f16(
          false, af, false, bf[j], (short)0, acc[i][j], false, false);
  }
}

// MMA step, A held as f32 in LDS (async-staged tile), convert while building frag
__device__ __forceinline__ void mma_k_f32A(const float (&As)[BM][SA32],
                                           const _Float16 (&Bs)[BK][SB],
                                           v8f (&acc)[4][2],
                                           int wr, int wc, int hh, int mr) {
  v16h bf[2];
  load_bfrag(Bs, bf, wc, hh, mr);
#pragma unroll
  for (int i = 0; i < 4; ++i) {
    v16h af;
    const int m = wr * 64 + i * 16 + mr;
#pragma unroll
    for (int p = 0; p < 8; ++p) {
      const int kb = frag_kb(p, hh);
      af[2 * p]     = (_Float16)As[m][kb];
      af[2 * p + 1] = (_Float16)As[m][kb + 1];
    }
#pragma unroll
    for (int j = 0; j < 2; ++j)
      acc[i][j] = __builtin_amdgcn_wmma_f32_16x16x32_f16(
          false, af, false, bf[j], (short)0, acc[i][j], false, false);
  }
}

// -------------------------- staging helpers --------------------------------
__device__ __forceinline__ void stage_A16(const float* __restrict__ A, size_t lda,
                                          int row0, int k0, int tid,
                                          _Float16 (&As)[BM][SA]) {
#pragma unroll
  for (int p = 0; p < 4; ++p) {
    const int r = (tid >> 3) + p * 32;   // 0..127
    const int c = (tid & 7) * 4;         // 0..28
    const float* gp = A + (size_t)(row0 + r) * lda + (size_t)(k0 + c);
    __builtin_prefetch(gp + BK, 0, 1);   // next k-tile -> global_prefetch_b8
    const float4 v = *(const float4*)gp;
    As[r][c + 0] = (_Float16)v.x;
    As[r][c + 1] = (_Float16)v.y;
    As[r][c + 2] = (_Float16)v.z;
    As[r][c + 3] = (_Float16)v.w;
  }
}

// async (VGPR-bypassing) f32 tile stage; fallback = LDS float4 stores
__device__ __forceinline__ void stage_A32(const float* __restrict__ A, size_t lda,
                                          int row0, int k0, int tid,
                                          float (&As)[BM][SA32]) {
#pragma unroll
  for (int p = 0; p < 4; ++p) {
    const int r = (tid >> 3) + p * 32;
    const int c = (tid & 7) * 4;
    const float* gp = A + (size_t)(row0 + r) * lda + (size_t)(k0 + c);
#if ASYNC_OK
    // AS-qualified pointer round-trip: low 32 bits of a generic LDS pointer are
    // the LDS offset; low 64 bits of a generic global pointer are the address.
    __builtin_amdgcn_global_load_async_to_lds_b128(
        (as1_v4i_p)(uintptr_t)gp, (as3_v4i_p)(uintptr_t)&As[r][c], 0, 0);
#else
    *(float4*)&As[r][c] = *(const float4*)gp;
#endif
  }
}

__device__ __forceinline__ void stage_B16(const float* __restrict__ B, size_t ldb,
                                          int k0, int col0, int tid,
                                          _Float16 (&Bs)[BK][SB]) {
#pragma unroll
  for (int p = 0; p < 4; ++p) {
    const int r = (tid >> 5) + p * 8;    // 0..31
    const int c = (tid & 31) * 4;        // 0..124
    const float* gp = B + (size_t)(k0 + r) * ldb + (size_t)(col0 + c);
    const float4 v = *(const float4*)gp;
    Bs[r][c + 0] = (_Float16)v.x;
    Bs[r][c + 1] = (_Float16)v.y;
    Bs[r][c + 2] = (_Float16)v.z;
    Bs[r][c + 3] = (_Float16)v.w;
  }
}

__device__ __forceinline__ void acc_zero(v8f (&acc)[4][2]) {
#pragma unroll
  for (int i = 0; i < 4; ++i)
#pragma unroll
    for (int j = 0; j < 2; ++j)
#pragma unroll
      for (int r = 0; r < 8; ++r) acc[i][j][r] = 0.0f;
}

// ---------------------------------------------------------------------------
// Feature GEMM: C = A @ B, A:[M,K] (lda=K), B:[K,Nc], small K. grid=(Nc/BN, M/BM)
// ---------------------------------------------------------------------------
__launch_bounds__(256)
__global__ void k_gemm(const float* __restrict__ A, const float* __restrict__ B,
                       float* __restrict__ C, int Nc, int K) {
  __shared__ _Float16 As[BM][SA];
  __shared__ _Float16 Bs[BK][SB];
  const int tid = threadIdx.x, lane = tid & 31, wid = tid >> 5;
  const int wr = wid >> 2, wc = wid & 3, hh = lane >> 4, mr = lane & 15;
  const int row0 = blockIdx.y * BM, col0 = blockIdx.x * BN;

  v8f acc[4][2];
  acc_zero(acc);
  for (int k0 = 0; k0 < K; k0 += BK) {
    stage_A16(A, (size_t)K, row0, k0, tid, As);
    stage_B16(B, (size_t)Nc, k0, col0, tid, Bs);
    __syncthreads();
    mma_k_f16(As, Bs, acc, wr, wc, hh, mr);
    __syncthreads();
  }
#pragma unroll
  for (int i = 0; i < 4; ++i)
#pragma unroll
    for (int j = 0; j < 2; ++j)
#pragma unroll
      for (int r = 0; r < 8; ++r) {
        const int row = row0 + wr * 64 + i * 16 + r + 8 * hh;
        const int col = col0 + wc * 32 + j * 16 + mr;
        C[(size_t)row * Nc + col] = acc[i][j][r];
      }
}

// ---------------------------------------------------------------------------
// Split-K adjacency GEMM: C += A[.,kz]@B[kz,.]  (atomic f32 accumulate).
// A:[M,8192] (lda = N_NODES), B:[8192,Nc].  grid=(Nc/BN, M/BM, SPLITK).
// Double-buffered LDS; A tile staged with GLOBAL_LOAD_ASYNC_TO_LDS_B128.
// ---------------------------------------------------------------------------
__launch_bounds__(256)
__global__ void k_gemm_sk(const float* __restrict__ A, const float* __restrict__ B,
                          float* __restrict__ C, int Nc) {
  __shared__ float    As[2][BM][SA32];   // 2 * 128*36*4  = 36 KB
  __shared__ _Float16 Bs[2][BK][SB];     // 2 * 32*136*2  = 17 KB
  const int tid = threadIdx.x, lane = tid & 31, wid = tid >> 5;
  const int wr = wid >> 2, wc = wid & 3, hh = lane >> 4, mr = lane & 15;
  const int row0 = blockIdx.y * BM, col0 = blockIdx.x * BN;
  const int kbase = blockIdx.z * KSLICE;

  v8f acc[4][2];
  acc_zero(acc);

  stage_A32(A, (size_t)N_NODES, row0, kbase, tid, As[0]);
  stage_B16(B, (size_t)Nc, kbase, col0, tid, Bs[0]);
  asm volatile("s_wait_asynccnt 0x0" ::: "memory");
  __syncthreads();

  const int ktiles = KSLICE / BK;        // 32
  for (int kt = 0; kt < ktiles; ++kt) {
    const int cur = kt & 1, nxt = cur ^ 1;
    if (kt + 1 < ktiles) {
      stage_A32(A, (size_t)N_NODES, row0, kbase + (kt + 1) * BK, tid, As[nxt]);
      stage_B16(B, (size_t)Nc, kbase + (kt + 1) * BK, col0, tid, Bs[nxt]);
    }
    mma_k_f32A(As[cur], Bs[cur], acc, wr, wc, hh, mr);
    asm volatile("s_wait_asynccnt 0x0" ::: "memory");
    __syncthreads();
  }

#pragma unroll
  for (int i = 0; i < 4; ++i)
#pragma unroll
    for (int j = 0; j < 2; ++j)
#pragma unroll
      for (int r = 0; r < 8; ++r) {
        const int row = row0 + wr * 64 + i * 16 + r + 8 * hh;
        const int col = col0 + wc * 32 + j * 16 + mr;
        atomicAdd(C + (size_t)row * Nc + col, acc[i][j][r]);
      }
}

// ---------------------------------------------------------------------------
// pert_adj: per graph b,  PA = pert_W[gid[b]] @ adj[b*256:(b+1)*256, :]
//           (+ pert_b[gid[b]] on the diagonal block), then (PA>0)?1:0
// grid = (8192/BN, 256/BM, 32)
// ---------------------------------------------------------------------------
__launch_bounds__(256)
__global__ void k_pert_adj(const float* __restrict__ pert_W,
                           const float* __restrict__ pert_b,
                           const int* __restrict__ graph_id,
                           const float* __restrict__ adj,
                           float* __restrict__ out) {
  __shared__ _Float16 As[BM][SA];
  __shared__ _Float16 Bs[BK][SB];
  const int b = blockIdx.z;
  const int g = graph_id[b];
  const float* A = pert_W + (size_t)g * C_DIM * C_DIM;   // [256,256]
  const float* B = adj + (size_t)b * C_DIM * N_NODES;    // [256,8192]

  const int tid = threadIdx.x, lane = tid & 31, wid = tid >> 5;
  const int wr = wid >> 2, wc = wid & 3, hh = lane >> 4, mr = lane & 15;
  const int row0 = blockIdx.y * BM;   // local row within graph
  const int col0 = blockIdx.x * BN;

  v8f acc[4][2];
  acc_zero(acc);
  for (int k0 = 0; k0 < C_DIM; k0 += BK) {
    stage_A16(A, (size_t)C_DIM, row0, k0, tid, As);
    stage_B16(B, (size_t)N_NODES, k0, col0, tid, Bs);
    __syncthreads();
    mma_k_f16(As, Bs, acc, wr, wc, hh, mr);
    __syncthreads();
  }

  const float* biasBlk = pert_b + (size_t)g * C_DIM * C_DIM;
#pragma unroll
  for (int i = 0; i < 4; ++i)
#pragma unroll
    for (int j = 0; j < 2; ++j)
#pragma unroll
      for (int r = 0; r < 8; ++r) {
        const int lrow = row0 + wr * 64 + i * 16 + r + 8 * hh;   // 0..255
        const int col  = col0 + wc * 32 + j * 16 + mr;           // 0..8191
        float v = acc[i][j][r];
        if ((col >> 8) == b) v += biasBlk[(size_t)lrow * C_DIM + (col & 255)];
        out[(size_t)(b * C_DIM + lrow) * N_NODES + col] = (v > 0.0f) ? 1.0f : 0.0f;
      }
}

// ---------------------------------------------------------------------------
__global__ void k_scatter(const int* __restrict__ ei, float* __restrict__ adj) {
  const int e = blockIdx.x * blockDim.x + threadIdx.x;
  if (e < E_EDGES) {
    const int r = ei[e];
    const int c = ei[E_EDGES + e];
    atomicAdd(adj + (size_t)r * N_NODES + c, 1.0f);
  }
}

__global__ void k_mask(const float* __restrict__ x, const float* __restrict__ mask_W,
                       const int* __restrict__ graph_id,
                       float* __restrict__ Mo, float* __restrict__ xm) {
  const int idx = blockIdx.x * 256 + threadIdx.x;
  if (idx < N_NODES * C_DIM) {
    const int node = idx >> 8;
    const int j    = idx & 255;
    const int b    = node >> 8;
    const int li   = node & 255;
    const int g    = graph_id[b];
    const float w  = mask_W[(size_t)g * C_DIM * C_DIM + (size_t)li * C_DIM + j];
    const float m  = (w > 0.0f) ? 1.0f : 0.0f;     // hard_where(sigmoid(w))
    Mo[idx] = m;
    xm[idx] = m * x[idx];
  }
}

__global__ void k_relu(const float* __restrict__ in, float* __restrict__ out, int n) {
  const int i = blockIdx.x * 256 + threadIdx.x;
  if (i < n) {
    const float v = in[i];
    out[i] = v > 0.0f ? v : 0.0f;
  }
}

// mean/max pool per 256-node graph stripe + 2-class MLP + sigmoid. block=128, grid=B
__launch_bounds__(128)
__global__ void k_pool_mlp(const float* __restrict__ h, const float* __restrict__ W,
                           const float* __restrict__ bias, float* __restrict__ pred) {
  __shared__ float red[NCLS_][H_DIM];
  const int b = blockIdx.x;
  const int f = threadIdx.x;
  float s = 0.0f, mx = -3.402823466e38f;
  const float* hp = h + (size_t)b * C_DIM * H_DIM + f;
  for (int r = 0; r < C_DIM; ++r) {
    const float v = hp[(size_t)r * H_DIM];
    s += v;
    mx = v > mx ? v : mx;
  }
  const float mean = s * (1.0f / (float)C_DIM);
#pragma unroll
  for (int c = 0; c < NCLS_; ++c)
    red[c][f] = mean * W[f * NCLS_ + c] + mx * W[(H_DIM + f) * NCLS_ + c];
  __syncthreads();
  for (int st = H_DIM / 2; st > 0; st >>= 1) {
    if (f < st) {
#pragma unroll
      for (int c = 0; c < NCLS_; ++c) red[c][f] += red[c][f + st];
    }
    __syncthreads();
  }
  if (f < NCLS_) {
    const float logit = red[f][0] + bias[f];
    pred[b * NCLS_ + f] = 1.0f / (1.0f + expf(-logit));
  }
}

// ---------------------------------------------------------------------------
static void adj_layer(const float* A, const float* hw, float* tmp, float* hout,
                      hipStream_t stream) {
  const size_t nh = (size_t)N_NODES * H_DIM;
  (void)hipMemsetAsync(tmp, 0, nh * sizeof(float), stream);
  k_gemm_sk<<<dim3(1, N_NODES / BM, SPLITK), 256, 0, stream>>>(A, hw, tmp, H_DIM);
  k_relu<<<(int)(nh / 256), 256, 0, stream>>>(tmp, hout, (int)nh);
}

static void run_predictor(const float* xin, int Cin, const float* A,
                          const float* w0, const float* w1, const float* w2,
                          const float* mlpW, const float* mlpb,
                          float* hw, float* hA, float* hB, float* tmp,
                          float* predOut, hipStream_t stream) {
  const dim3 blk(256);
  const dim3 g64(1, N_NODES / BM);
  k_gemm<<<g64, blk, 0, stream>>>(xin, w0, hw, H_DIM, Cin);
  adj_layer(A, hw, tmp, hA, stream);
  k_gemm<<<g64, blk, 0, stream>>>(hA, w1, hw, H_DIM, H_DIM);
  adj_layer(A, hw, tmp, hB, stream);
  k_gemm<<<g64, blk, 0, stream>>>(hB, w2, hw, H_DIM, H_DIM);
  adj_layer(A, hw, tmp, hA, stream);
  k_pool_mlp<<<B_G, H_DIM, 0, stream>>>(hA, mlpW, mlpb, predOut);
}

extern "C" void kernel_launch(void* const* d_in, const int* in_sizes, int n_in,
                              void* d_out, int out_size, void* d_ws, size_t ws_size,
                              hipStream_t stream) {
  (void)in_sizes; (void)n_in; (void)out_size; (void)ws_size;
  const float* x      = (const float*)d_in[0];
  const int*   ei     = (const int*)d_in[1];
  const int*   gid    = (const int*)d_in[2];
  /* d_in[3] = batch, implicit in indexing */
  const float* pert_W = (const float*)d_in[4];
  const float* pert_b = (const float*)d_in[5];
  const float* mask_W = (const float*)d_in[6];
  const float* w0     = (const float*)d_in[7];
  const float* w1     = (const float*)d_in[8];
  const float* w2     = (const float*)d_in[9];
  const float* mlpW   = (const float*)d_in[10];
  const float* mlpb   = (const float*)d_in[11];

  float* out = (float*)d_out;
  const size_t NN = (size_t)N_NODES * N_NODES;
  float* adj  = out;
  float* padj = out + NN;
  float* Mo   = out + 2 * NN;
  float* pred = Mo + (size_t)N_NODES * C_DIM;
  float* aug  = pred + B_G * NCLS_;
  float* xm   = aug + B_G * NCLS_;

  float* hw  = (float*)d_ws;
  float* hA  = hw  + (size_t)N_NODES * H_DIM;
  float* hB  = hA  + (size_t)N_NODES * H_DIM;
  float* tmp = hB  + (size_t)N_NODES * H_DIM;

  (void)hipMemsetAsync(adj, 0, NN * sizeof(float), stream);
  k_scatter<<<E_EDGES / 256, 256, 0, stream>>>(ei, adj);
  k_mask<<<(N_NODES * C_DIM) / 256, 256, 0, stream>>>(x, mask_W, gid, Mo, xm);
  k_pert_adj<<<dim3(N_NODES / BN, C_DIM / BM, B_G), 256, 0, stream>>>(
      pert_W, pert_b, gid, adj, padj);

  run_predictor(x,  C_DIM, adj,  w0, w1, w2, mlpW, mlpb, hw, hA, hB, tmp, pred, stream);
  run_predictor(xm, C_DIM, padj, w0, w1, w2, mlpW, mlpb, hw, hA, hB, tmp, aug,  stream);
}
